// MultiHeadAttention_91027536872189
// MI455X (gfx1250) — matrix operand order repaired and checked by reading, not compiled, not run on hardware
//
#include <hip/hip_runtime.h>

// ---------------------------------------------------------------------------
// MHA forward for gfx1250 (MI455X): bf16 WMMA GEMMs + flash-attention core,
// with CDNA5 async global->LDS copies (ASYNCcnt) and LDS double buffering.
// B=4, S=2048, D=1024, H=16, HD=64. fp32 in/out, bf16 tensor-core math.
// ---------------------------------------------------------------------------

#define Bn  4
#define Sn  2048
#define Dn  1024
#define Hn  16
#define HDn 64
#define Mn  (Bn * Sn)   // 8192 rows for the projection GEMMs

typedef __attribute__((ext_vector_type(16))) __bf16 v16bf;
typedef __attribute__((ext_vector_type(8)))  float  v8f;

union BF16x16 { unsigned u[8]; v16bf v; };

__device__ __forceinline__ unsigned short f32_to_bf16(float f) {
  union { float f; unsigned u; } c; c.f = f;
  unsigned u = c.u;
  u += 0x7FFFu + ((u >> 16) & 1u);   // round-to-nearest-even
  return (unsigned short)(u >> 16);
}

// ---------------------------------------------------------------------------
// CDNA5 async copy: GLOBAL_LOAD_ASYNC_TO_LDS_B128 (memory -> LDS, ASYNCcnt).
// LDS destination is a per-lane byte offset (low 32 bits of the flat shared
// address are the LDS offset per the aperture mapping).
// ---------------------------------------------------------------------------
__device__ __forceinline__ unsigned lds_off32(const void* p) {
  return (unsigned)(size_t)p;
}

__device__ __forceinline__ void async_copy_b128(unsigned lds_off, const void* gaddr) {
  asm volatile("global_load_async_to_lds_b128 %0, %1, off"
               :: "v"(lds_off), "v"(gaddr)
               : "memory");
}

__device__ __forceinline__ void wait_async0() {
#if __has_builtin(__builtin_amdgcn_s_wait_asynccnt)
  __builtin_amdgcn_s_wait_asynccnt(0);
#else
  asm volatile("s_wait_asynccnt 0" ::: "memory");
#endif
}

// ---------------------------------------------------------------------------
// 1) x fp32 -> bf16, 4 elements / thread
// ---------------------------------------------------------------------------
__global__ __launch_bounds__(256) void cvt_f32_bf16_kernel(
    const float* __restrict__ in, unsigned short* __restrict__ out, int n) {
  int i = (blockIdx.x * 256 + threadIdx.x) * 4;
  if (i + 3 >= n) return;
  float4 f = *(const float4*)(in + i);
  unsigned short t[4];
  t[0] = f32_to_bf16(f.x); t[1] = f32_to_bf16(f.y);
  t[2] = f32_to_bf16(f.z); t[3] = f32_to_bf16(f.w);
  *(uint2*)(out + i) = *(const uint2*)t;
}

// ---------------------------------------------------------------------------
// 2) W [K][N] fp32 -> Wt [N][K] bf16 (so B-operand tiles are row-major over K)
// ---------------------------------------------------------------------------
__global__ __launch_bounds__(256) void wtrans_kernel(
    const float* __restrict__ Wq, const float* __restrict__ Wk,
    const float* __restrict__ Wv, const float* __restrict__ Wo,
    unsigned short* __restrict__ Wqt, unsigned short* __restrict__ Wkt,
    unsigned short* __restrict__ Wvt, unsigned short* __restrict__ Wot) {
  const float* src = (blockIdx.z == 0) ? Wq : (blockIdx.z == 1) ? Wk
                   : (blockIdx.z == 2) ? Wv : Wo;
  unsigned short* dst = (blockIdx.z == 0) ? Wqt : (blockIdx.z == 1) ? Wkt
                      : (blockIdx.z == 2) ? Wvt : Wot;
  __shared__ float tile[32][33];
  const int n0 = blockIdx.x * 32, k0 = blockIdx.y * 32;
  const int tx = threadIdx.x, ty = threadIdx.y;   // block (32, 8)
  #pragma unroll
  for (int dy = 0; dy < 32; dy += 8)
    tile[ty + dy][tx] = src[(size_t)(k0 + ty + dy) * Dn + n0 + tx];
  __syncthreads();
  #pragma unroll
  for (int dy = 0; dy < 32; dy += 8)
    dst[(size_t)(n0 + ty + dy) * Dn + k0 + tx] = f32_to_bf16(tile[tx][ty + dy]);
}

// ---------------------------------------------------------------------------
// WMMA fragment gathers from LDS tiles stored row-major with stride 40 bf16.
//   A 16x32 bf16: lane%16 = M, k = (v>=4?16:0) + 8*(lane/16) + 2*(v&3)
//   B 32x16 bf16: lane%16 = N, k = (v>=4?16:0) + 8*(lane/16) + 2*(v&3)
// Both gathers are two contiguous 16B runs -> compiler emits ds_load_b128.
// ---------------------------------------------------------------------------
#define LSTR  40               // LDS row stride in bf16 elements (32 + 8 pad)
#define GTILE (128 * LSTR)     // one GEMM LDS tile in bf16 elements

__device__ __forceinline__ v16bf load_afrag(const unsigned short* lds, int row,
                                            int half, int l15) {
  BF16x16 f;
  #pragma unroll
  for (int v = 0; v < 8; ++v) {
    int k = ((v >= 4) ? 16 : 0) + half * 8 + (v & 3) * 2;
    f.u[v] = *(const unsigned*)&lds[(row + l15) * LSTR + k];
  }
  return f.v;
}
__device__ __forceinline__ v16bf load_bfrag(const unsigned short* lds, int row,
                                            int half, int l15) {
  BF16x16 f;
  #pragma unroll
  for (int v = 0; v < 8; ++v) {
    int k = ((v >= 4) ? 16 : 0) + half * 8 + (v & 3) * 2;
    f.u[v] = *(const unsigned*)&lds[(row + l15) * LSTR + k];
  }
  return f.v;
}

// ---------------------------------------------------------------------------
// Shared pipelined GEMM core: acc[4][2] += A[128,1024] x Bt[128,1024]^T tile.
// Async global->LDS fills, double-buffered LDS, one wait+barrier per k-step.
// ---------------------------------------------------------------------------
__device__ __forceinline__ void gemm_core(
    const unsigned short* __restrict__ Aglob,
    const unsigned short* __restrict__ Bglob,
    unsigned short* Alds,   // [2][GTILE]
    unsigned short* Blds,   // [2][GTILE]
    int m_base, int n_base, int tid,
    int wm, int wn, int half, int l15, v8f acc[4][2]) {

  const unsigned aoff0 = lds_off32(Alds), boff0 = lds_off32(Blds);

  auto issue_tile = [&](int buf, int k0) {
    const unsigned ab = aoff0 + (unsigned)(buf * GTILE * 2);
    const unsigned bb = boff0 + (unsigned)(buf * GTILE * 2);
    #pragma unroll
    for (int it = 0; it < 2; ++it) {          // 256 thr x 2 x 16B = 8KB/tile
      int i = tid + it * 256;
      int r = i >> 2, c = (i & 3) << 3;
      unsigned lo = (unsigned)((r * LSTR + c) * 2);
      async_copy_b128(ab + lo, Aglob + (size_t)(m_base + r) * Dn + k0 + c);
      async_copy_b128(bb + lo, Bglob + (size_t)(n_base + r) * Dn + k0 + c);
    }
  };

  issue_tile(0, 0);
  wait_async0();
  __syncthreads();

  for (int kt = 0; kt < Dn / 32; ++kt) {
    const int cur = kt & 1;
    if (kt + 1 < Dn / 32) issue_tile(cur ^ 1, (kt + 1) * 32);

    const unsigned short* Al = Alds + cur * GTILE;
    const unsigned short* Bl = Blds + cur * GTILE;
    v16bf afr[4], bfr[2];
    #pragma unroll
    for (int mi = 0; mi < 4; ++mi)
      afr[mi] = load_afrag(Al, wm * 64 + mi * 16, half, l15);
    #pragma unroll
    for (int ni = 0; ni < 2; ++ni)
      bfr[ni] = load_bfrag(Bl, wn * 32 + ni * 16, half, l15);
    #pragma unroll
    for (int mi = 0; mi < 4; ++mi)
      #pragma unroll
      for (int ni = 0; ni < 2; ++ni)
        acc[mi][ni] = __builtin_amdgcn_wmma_f32_16x16x32_bf16(
            false, afr[mi], false, bfr[ni], (short)0, acc[mi][ni], false, false);

    wait_async0();          // next buffer's fill complete (this wave)
    __syncthreads();        // ... and for all waves
  }
}

// ---------------------------------------------------------------------------
// 3) QKV projection GEMM. Block tile 128x128, kc=32, 8 waves (2Mx4N).
//    mode (blockIdx.z): 0=Q (scaled 1/8, head-split), 1=K (head-split),
//                       2=V (stored transposed [B,H,HD,S]).
// ---------------------------------------------------------------------------
__global__ __launch_bounds__(256) void qkv_gemm_kernel(
    const unsigned short* __restrict__ xb,
    const unsigned short* __restrict__ Wqt, const unsigned short* __restrict__ Wkt,
    const unsigned short* __restrict__ Wvt,
    const float* __restrict__ bq, const float* __restrict__ bk,
    const float* __restrict__ bv,
    unsigned short* __restrict__ Qb, unsigned short* __restrict__ Kb,
    unsigned short* __restrict__ Vtb) {
  const int mode = blockIdx.z;
  const unsigned short* Wt = (mode == 0) ? Wqt : (mode == 1) ? Wkt : Wvt;
  const float* bias = (mode == 0) ? bq : (mode == 1) ? bk : bv;

  const int m_base = blockIdx.y * 128;
  const int n_base = blockIdx.x * 128;
  const int tid = threadIdx.x;
  const int lane = tid & 31, wave = tid >> 5;
  const int wm = wave >> 2, wn = wave & 3;
  const int half = lane >> 4, l15 = lane & 15;

  __shared__ unsigned short Alds[2 * GTILE];
  __shared__ unsigned short Blds[2 * GTILE];

  v8f acc[4][2];
  #pragma unroll
  for (int mi = 0; mi < 4; ++mi)
    #pragma unroll
    for (int ni = 0; ni < 2; ++ni)
      #pragma unroll
      for (int e = 0; e < 8; ++e) acc[mi][ni][e] = 0.f;

  gemm_core(xb, Wt, Alds, Blds, m_base, n_base, tid, wm, wn, half, l15, acc);

  const float scale = (mode == 0) ? 0.125f : 1.0f;   // 1/sqrt(HD) folded into Q
  #pragma unroll
  for (int mi = 0; mi < 4; ++mi)
    #pragma unroll
    for (int ni = 0; ni < 2; ++ni)
      #pragma unroll
      for (int r = 0; r < 8; ++r) {
        int m = m_base + wm * 64 + mi * 16 + r + half * 8;
        int n = n_base + wn * 32 + ni * 16 + l15;
        float val = (acc[mi][ni][r] + bias[n]) * scale;
        int b = m >> 11, s = m & (Sn - 1);
        int h = n >> 6, hd = n & (HDn - 1);
        unsigned short bv16 = f32_to_bf16(val);
        if (mode == 2)
          Vtb[((size_t)(b * Hn + h) * HDn + hd) * Sn + s] = bv16;
        else {
          unsigned short* dst = (mode == 0) ? Qb : Kb;
          dst[((size_t)(b * Hn + h) * Sn + s) * HDn + hd] = bv16;
        }
      }
}

// ---------------------------------------------------------------------------
// 4) Flash attention: grid (S/64, B*H), 4 waves x 16 query rows.
//    Online softmax; K/Vt tiles async-streamed into double-buffered LDS.
// ---------------------------------------------------------------------------
__global__ __launch_bounds__(128) void attn_kernel(
    const unsigned short* __restrict__ Qb, const unsigned short* __restrict__ Kb,
    const unsigned short* __restrict__ Vtb, unsigned short* __restrict__ Ob) {
  const int bh = blockIdx.y;                 // 0..63
  const int b = bh >> 4, h = bh & 15;
  const int qt = blockIdx.x;                 // query tile, 0..31
  const int lane = threadIdx.x & 31, wave = threadIdx.x >> 5;
  const int half = lane >> 4, l15 = lane & 15;

  #define KSTR  72                           // 64 + 8 pad, bf16 elements
  #define KTILE (64 * KSTR)
  __shared__ unsigned short Klds[2 * KTILE];
  __shared__ unsigned short Vlds[2 * KTILE];  // Vt tile: row = d, col = key
  __shared__ unsigned short Plds[4][16 * KSTR];

  const unsigned short* Kbase = Kb + (size_t)bh * Sn * HDn;
  const unsigned short* Vbase = Vtb + (size_t)bh * HDn * Sn;
  const unsigned koff0 = lds_off32(Klds), voff0 = lds_off32(Vlds);

  auto issue_tile = [&](int buf, int t) {
    const unsigned kb = koff0 + (unsigned)(buf * KTILE * 2);
    const unsigned vb = voff0 + (unsigned)(buf * KTILE * 2);
    #pragma unroll
    for (int it = 0; it < 4; ++it) {          // 128 thr x 4 x 16B = 8KB/tile
      int i = threadIdx.x + it * 128;
      int r = i >> 3, c = (i & 7) << 3;
      unsigned lo = (unsigned)((r * KSTR + c) * 2);
      async_copy_b128(kb + lo, Kbase + (size_t)(t * 64 + r) * HDn + c);
      async_copy_b128(vb + lo, Vbase + (size_t)r * Sn + t * 64 + c);
    }
  };

  // Q fragments for this wave's 16 rows, K = 0..63 (2 A-frags), kept resident.
  const unsigned short* Qbase = Qb + ((size_t)bh * Sn + qt * 64) * HDn;
  v16bf qfr[2];
  #pragma unroll
  for (int kk = 0; kk < 2; ++kk) {
    BF16x16 f;
    #pragma unroll
    for (int v = 0; v < 8; ++v) {
      int k = kk * 32 + ((v >= 4) ? 16 : 0) + half * 8 + (v & 3) * 2;
      f.u[v] = *(const unsigned*)(Qbase + (size_t)(wave * 16 + l15) * HDn + k);
    }
    qfr[kk] = f.v;
  }

  v8f oacc[4];
  #pragma unroll
  for (int di = 0; di < 4; ++di)
    #pragma unroll
    for (int e = 0; e < 8; ++e) oacc[di][e] = 0.f;
  float mrun[8], lrun[8];
  #pragma unroll
  for (int r = 0; r < 8; ++r) { mrun[r] = -3.0e38f; lrun[r] = 0.f; }

  issue_tile(0, 0);
  wait_async0();
  __syncthreads();

  for (int t = 0; t <= qt; ++t) {
    const int cur = t & 1;
    if (t < qt) issue_tile(cur ^ 1, t + 1);
    const unsigned short* Kl = Klds + cur * KTILE;
    const unsigned short* Vl = Vlds + cur * KTILE;

    // ---- scores S = Q . K^T  (16 x 64 per wave) ----
    v8f sacc[4];
    #pragma unroll
    for (int ni = 0; ni < 4; ++ni) {
      #pragma unroll
      for (int e = 0; e < 8; ++e) sacc[ni][e] = 0.f;
      #pragma unroll
      for (int kk = 0; kk < 2; ++kk) {
        BF16x16 f;
        #pragma unroll
        for (int v = 0; v < 8; ++v) {
          int k = kk * 32 + ((v >= 4) ? 16 : 0) + half * 8 + (v & 3) * 2;
          f.u[v] = *(const unsigned*)&Kl[(ni * 16 + l15) * KSTR + k];
        }
        sacc[ni] = __builtin_amdgcn_wmma_f32_16x16x32_bf16(
            false, qfr[kk], false, f.v, (short)0, sacc[ni], false, false);
      }
    }

    // ---- causal mask on the diagonal tile only ----
    if (t == qt) {
      #pragma unroll
      for (int ni = 0; ni < 4; ++ni)
        #pragma unroll
        for (int r = 0; r < 8; ++r) {
          int qg = wave * 16 + r + half * 8;       // row within this 64-tile
          int kg = ni * 16 + l15;                  // key within this 64-tile
          if (kg > qg) sacc[ni][r] = -1.0e30f;
        }
    }

    // ---- online softmax, per C-fragment row (m = r + 8*half) ----
    float alpha[8];
    #pragma unroll
    for (int r = 0; r < 8; ++r) {
      float mx = sacc[0][r];
      #pragma unroll
      for (int ni = 1; ni < 4; ++ni) mx = fmaxf(mx, sacc[ni][r]);
      #pragma unroll
      for (int off = 8; off; off >>= 1) mx = fmaxf(mx, __shfl_xor(mx, off, 16));
      float mnew = fmaxf(mrun[r], mx);
      alpha[r] = __expf(mrun[r] - mnew);
      float rsum = 0.f;
      #pragma unroll
      for (int ni = 0; ni < 4; ++ni) {
        float p = __expf(sacc[ni][r] - mnew);
        sacc[ni][r] = p;
        rsum += p;
      }
      #pragma unroll
      for (int off = 8; off; off >>= 1) rsum += __shfl_xor(rsum, off, 16);
      lrun[r] = lrun[r] * alpha[r] + rsum;
      mrun[r] = mnew;
    }
    #pragma unroll
    for (int di = 0; di < 4; ++di)
      #pragma unroll
      for (int r = 0; r < 8; ++r) oacc[di][r] *= alpha[r];

    // ---- P: C-frag -> bf16 A-frag via wave-private LDS ----
    #pragma unroll
    for (int ni = 0; ni < 4; ++ni)
      #pragma unroll
      for (int r = 0; r < 8; ++r)
        Plds[wave][(r + half * 8) * KSTR + ni * 16 + l15] = f32_to_bf16(sacc[ni][r]);
    __syncthreads();                           // order P write -> P read

    v16bf pfr[2];
    #pragma unroll
    for (int kk = 0; kk < 2; ++kk) {
      BF16x16 f;
      #pragma unroll
      for (int v = 0; v < 8; ++v) {
        int k = kk * 32 + ((v >= 4) ? 16 : 0) + half * 8 + (v & 3) * 2;
        f.u[v] = *(const unsigned*)&Plds[wave][l15 * KSTR + k];
      }
      pfr[kk] = f.v;
    }
    // ---- O += P @ V (B-operand rows are d, reduction over keys) ----
    #pragma unroll
    for (int di = 0; di < 4; ++di)
      #pragma unroll
      for (int kk = 0; kk < 2; ++kk) {
        BF16x16 f;
        #pragma unroll
        for (int v = 0; v < 8; ++v) {
          int k = kk * 32 + ((v >= 4) ? 16 : 0) + half * 8 + (v & 3) * 2;
          f.u[v] = *(const unsigned*)&Vl[(di * 16 + l15) * KSTR + k];
        }
        oacc[di] = __builtin_amdgcn_wmma_f32_16x16x32_bf16(
            false, pfr[kk], false, f.v, (short)0, oacc[di], false, false);
      }

    wait_async0();          // next K/V buffer landed (this wave)
    __syncthreads();        // ... for all waves; also frees Plds
  }

  // ---- normalize and store to [B, S, D] bf16 ----
  float invl[8];
  #pragma unroll
  for (int r = 0; r < 8; ++r) invl[r] = 1.0f / lrun[r];
  #pragma unroll
  for (int di = 0; di < 4; ++di)
    #pragma unroll
    for (int r = 0; r < 8; ++r) {
      int sq = qt * 64 + wave * 16 + r + half * 8;
      Ob[((size_t)b * Sn + sq) * Dn + h * HDn + di * 16 + l15] =
          f32_to_bf16(oacc[di][r] * invl[r]);
    }
}

// ---------------------------------------------------------------------------
// 5) Output projection: out[8192,1024] = Ob @ Wot^T + bo (fp32 store)
// ---------------------------------------------------------------------------
__global__ __launch_bounds__(256) void oproj_gemm_kernel(
    const unsigned short* __restrict__ Ab, const unsigned short* __restrict__ Wt,
    const float* __restrict__ bo, float* __restrict__ out) {
  const int m_base = blockIdx.y * 128;
  const int n_base = blockIdx.x * 128;
  const int tid = threadIdx.x;
  const int lane = tid & 31, wave = tid >> 5;
  const int wm = wave >> 2, wn = wave & 3;
  const int half = lane >> 4, l15 = lane & 15;

  __shared__ unsigned short Alds[2 * GTILE];
  __shared__ unsigned short Blds[2 * GTILE];

  v8f acc[4][2];
  #pragma unroll
  for (int mi = 0; mi < 4; ++mi)
    #pragma unroll
    for (int ni = 0; ni < 2; ++ni)
      #pragma unroll
      for (int e = 0; e < 8; ++e) acc[mi][ni][e] = 0.f;

  gemm_core(Ab, Wt, Alds, Blds, m_base, n_base, tid, wm, wn, half, l15, acc);

  #pragma unroll
  for (int mi = 0; mi < 4; ++mi)
    #pragma unroll
    for (int ni = 0; ni < 2; ++ni)
      #pragma unroll
      for (int r = 0; r < 8; ++r) {
        int m = m_base + wm * 64 + mi * 16 + r + half * 8;
        int n = n_base + wn * 32 + ni * 16 + l15;
        out[(size_t)m * Dn + n] = acc[mi][ni][r] + bo[n];
      }
}

// ---------------------------------------------------------------------------
extern "C" void kernel_launch(void* const* d_in, const int* in_sizes, int n_in,
                              void* d_out, int out_size, void* d_ws, size_t ws_size,
                              hipStream_t stream) {
  (void)in_sizes; (void)n_in; (void)out_size; (void)ws_size;
  const float* x  = (const float*)d_in[0];
  const float* Wq = (const float*)d_in[1];
  const float* bq = (const float*)d_in[2];
  const float* Wk = (const float*)d_in[3];
  const float* bk = (const float*)d_in[4];
  const float* Wv = (const float*)d_in[5];
  const float* bv = (const float*)d_in[6];
  const float* Wo = (const float*)d_in[7];
  const float* bo = (const float*)d_in[8];
  float* out = (float*)d_out;

  char* ws = (char*)d_ws;
  const size_t MB = 1ull << 20;
  unsigned short* xb  = (unsigned short*)(ws + 0);        // 16 MiB
  unsigned short* Wqt = (unsigned short*)(ws + 16 * MB);  // 2 MiB
  unsigned short* Wkt = (unsigned short*)(ws + 18 * MB);
  unsigned short* Wvt = (unsigned short*)(ws + 20 * MB);
  unsigned short* Wot = (unsigned short*)(ws + 22 * MB);
  unsigned short* Qb  = (unsigned short*)(ws + 24 * MB);  // 16 MiB [B,H,S,HD]
  unsigned short* Kb  = (unsigned short*)(ws + 40 * MB);  // 16 MiB [B,H,S,HD]
  unsigned short* Vtb = (unsigned short*)(ws + 56 * MB);  // 16 MiB [B,H,HD,S]
  unsigned short* Ob  = (unsigned short*)(ws + 72 * MB);  // 16 MiB [B,S,D]

  cvt_f32_bf16_kernel<<<(Mn * Dn) / 1024, 256, 0, stream>>>(x, xb, Mn * Dn);
  wtrans_kernel<<<dim3(32, 32, 4), dim3(32, 8), 0, stream>>>(
      Wq, Wk, Wv, Wo, Wqt, Wkt, Wvt, Wot);
  qkv_gemm_kernel<<<dim3(Dn / 128, Mn / 128, 3), 256, 0, stream>>>(
      xb, Wqt, Wkt, Wvt, bq, bk, bv, Qb, Kb, Vtb);
  attn_kernel<<<dim3(Sn / 64, Bn * Hn), 128, 0, stream>>>(Qb, Kb, Vtb, Ob);
  oproj_gemm_kernel<<<dim3(Dn / 128, Mn / 128), 256, 0, stream>>>(Ob, Wot, bo, out);
}
